// Encoder_72954314489957
// MI455X (gfx1250) — matrix operand order repaired and checked by reading, not compiled, not run on hardware
//
#include <hip/hip_runtime.h>

typedef __attribute__((ext_vector_type(16))) __bf16 v16bf;
typedef __attribute__((ext_vector_type(8)))  float  v8f;

#define N_NODES 50000
#define N_EDGES 800000
#define N_GRAPH 512
#define N_HEAD  4
#define N_CHAN  64
#define D_IN    128
#define D_ENC   256
#define HC      256                 // H*C
#define ET      (N_EDGES + N_NODES) // edges + self loops = 850000
#define D_CAT   (D_ENC + N_CHAN)    // 320

// ---- monotonic float<->uint encoding for atomicMax on floats -------------
__device__ __forceinline__ unsigned encf(float f) {
    unsigned u = __float_as_uint(f);
    return (u & 0x80000000u) ? ~u : (u | 0x80000000u);
}
__device__ __forceinline__ float decf(unsigned u) {
    u = (u & 0x80000000u) ? (u & 0x7FFFFFFFu) : ~u;
    return __uint_as_float(u);
}

__device__ __forceinline__ __bf16 bf_hi(float f) { return (__bf16)f; }
__device__ __forceinline__ __bf16 bf_lo(float f, __bf16 hi) {
    return (__bf16)(f - (float)hi);
}

// ---- generic f32-accurate WMMA GEMM:  Out[M,Nc] = A[M,K] @ W[K,Nc] + bias -
// one wave per 16x16 output tile; bf16 hi/lo split, f32 accumulate.
__global__ void gatv2_wmma_gemm(const float* __restrict__ A,
                                const float* __restrict__ W,
                                const float* __restrict__ bias,
                                float* __restrict__ Out,
                                int K, int Ncols, int tilesN, int totalTiles) {
    int wid = blockIdx.x * (blockDim.x >> 5) + (threadIdx.x >> 5);
    if (wid >= totalTiles) return;            // wave-uniform: EXEC stays all-ones
    int lane  = threadIdx.x & 31;
    int group = lane >> 4;                    // 0 or 1
    int idx15 = lane & 15;
    int tm = wid / tilesN, tn = wid % tilesN;
    int rowbase = tm * 16;
    int col     = tn * 16 + idx15;            // this lane's output column

    float bcol = bias[col];
    v8f acc;
#pragma unroll
    for (int r = 0; r < 8; ++r) acc[r] = bcol;

    const float* arow = A + (size_t)(rowbase + idx15) * K;

    for (int ks = 0; ks < K; ks += 32) {
        v16bf ahi, alo, bhi, blo;
#pragma unroll
        for (int v = 0; v < 8; ++v) {
            // A (16-bit 16x32): lanes hold row M=idx15; K = 16*(v>=4) + 8*group + 2*(v&3)
            int ka = ks + ((v & 4) ? 16 : 0) + 8 * group + 2 * (v & 3);
            float a0 = arow[ka];
            float a1 = arow[ka + 1];
            __bf16 a0h = bf_hi(a0), a1h = bf_hi(a1);
            ahi[2 * v]     = a0h;  alo[2 * v]     = bf_lo(a0, a0h);
            ahi[2 * v + 1] = a1h;  alo[2 * v + 1] = bf_lo(a1, a1h);
            // B (16-bit 32x16): lanes hold col N=idx15; K = 16*(v>=4) + 8*group + 2*(v&3)
            int kb = ks + ((v & 4) ? 16 : 0) + 8 * group + 2 * (v & 3);
            float b0 = W[(size_t)kb * Ncols + col];
            float b1 = W[(size_t)(kb + 1) * Ncols + col];
            __bf16 b0h = bf_hi(b0), b1h = bf_hi(b1);
            bhi[2 * v]     = b0h;  blo[2 * v]     = bf_lo(b0, b0h);
            bhi[2 * v + 1] = b1h;  blo[2 * v + 1] = bf_lo(b1, b1h);
        }
        acc = __builtin_amdgcn_wmma_f32_16x16x32_bf16(false, ahi, false, bhi, (short)0, acc, false, false);
        acc = __builtin_amdgcn_wmma_f32_16x16x32_bf16(false, ahi, false, blo, (short)0, acc, false, false);
        acc = __builtin_amdgcn_wmma_f32_16x16x32_bf16(false, alo, false, bhi, (short)0, acc, false, false);
    }
#pragma unroll
    for (int r = 0; r < 8; ++r)
        Out[(size_t)(rowbase + r + 8 * group) * Ncols + col] = acc[r];
}

// ---- per-edge attention logits + segment max (wave per edge) --------------
__global__ void gatv2_edge_logits(const int* __restrict__ ei,
                                  const float* __restrict__ xl,
                                  const float* __restrict__ xr,
                                  const float* __restrict__ att,
                                  float* __restrict__ ebuf,
                                  unsigned* __restrict__ menc) {
    int edge = blockIdx.x * (blockDim.x >> 5) + (threadIdx.x >> 5);
    if (edge >= ET) return;
    int lane = threadIdx.x & 31;
    int src, dst;
    if (edge < N_EDGES) { src = ei[edge]; dst = ei[N_EDGES + edge]; }
    else                { src = dst = edge - N_EDGES; }
    const float* pl = xl + (size_t)src * HC;
    const float* pr = xr + (size_t)dst * HC;
    float partial = 0.f;
#pragma unroll
    for (int j = 0; j < 8; ++j) {
        int c = lane * 8 + j;                 // lanes 0-7 -> head0, 8-15 -> head1, ...
        float s = pl[c] + pr[c];
        s = s > 0.f ? s : 0.2f * s;           // leaky_relu(·, 0.2)
        partial += s * att[c];                // att flat [H*C]
    }
    partial += __shfl_xor(partial, 4, 32);
    partial += __shfl_xor(partial, 2, 32);
    partial += __shfl_xor(partial, 1, 32);
    if ((lane & 7) == 0) {
        int h = lane >> 3;
        ebuf[(size_t)edge * N_HEAD + h] = partial;
        atomicMax(&menc[(size_t)dst * N_HEAD + h], encf(partial));
    }
}

// ---- softmax denominator: atomicAdd(exp(e - m[dst])) ----------------------
__global__ void gatv2_edge_denom(const int* __restrict__ ei,
                                 const float* __restrict__ ebuf,
                                 const unsigned* __restrict__ menc,
                                 float* __restrict__ denom) {
    int t = blockIdx.x * blockDim.x + threadIdx.x;
    if (t >= ET * N_HEAD) return;
    int edge = t >> 2, h = t & 3;
    int dst = (edge < N_EDGES) ? ei[N_EDGES + edge] : (edge - N_EDGES);
    float a = __expf(ebuf[t] - decf(menc[(size_t)dst * N_HEAD + h]));
    atomicAdd(&denom[(size_t)dst * N_HEAD + h], a);
}

// ---- weighted aggregation, head-merged: acc[dst,c] += sum_h alpha_h*xl[src,h*64+c]
__global__ void gatv2_edge_aggregate(const int* __restrict__ ei,
                                     const float* __restrict__ ebuf,
                                     const unsigned* __restrict__ menc,
                                     const float* __restrict__ denom,
                                     const float* __restrict__ xl,
                                     float* __restrict__ acc) {
    int edge = blockIdx.x * (blockDim.x >> 5) + (threadIdx.x >> 5);
    if (edge >= ET) return;
    int lane = threadIdx.x & 31;
    int src, dst;
    if (edge < N_EDGES) { src = ei[edge]; dst = ei[N_EDGES + edge]; }
    else                { src = dst = edge - N_EDGES; }
    float alpha = 0.f;
    if (lane < N_HEAD) {
        float m = decf(menc[(size_t)dst * N_HEAD + lane]);
        float a = __expf(ebuf[(size_t)edge * N_HEAD + lane] - m);
        alpha = a / (denom[(size_t)dst * N_HEAD + lane] + 1e-16f);
    }
    float a0 = __shfl(alpha, 0, 32);
    float a1 = __shfl(alpha, 1, 32);
    float a2 = __shfl(alpha, 2, 32);
    float a3 = __shfl(alpha, 3, 32);
    const float* pl = xl + (size_t)src * HC;
    float v0 = a0 * pl[lane]       + a1 * pl[64 + lane]
             + a2 * pl[128 + lane] + a3 * pl[192 + lane];
    float v1 = a0 * pl[32 + lane]  + a1 * pl[96 + lane]
             + a2 * pl[160 + lane] + a3 * pl[224 + lane];
    atomicAdd(&acc[(size_t)dst * N_CHAN + lane],      v0);
    atomicAdd(&acc[(size_t)dst * N_CHAN + 32 + lane], v1);
}

// ---- node finalize (head mean + bias + leaky 0.01) and graph pooling ------
__global__ void gatv2_node_pool(const float* __restrict__ acc,
                                const float* __restrict__ bias_gnn,
                                const int* __restrict__ batch,
                                float* __restrict__ gsum,
                                float* __restrict__ gcnt) {
    int t = blockIdx.x * blockDim.x + threadIdx.x;
    if (t >= N_NODES * N_CHAN) return;
    int n = t >> 6, c = t & 63;
    float hv = acc[t] * (1.0f / N_HEAD) + bias_gnn[c];
    hv = hv > 0.f ? hv : 0.01f * hv;
    int b = batch[n];
    atomicAdd(&gsum[b * N_CHAN + c], hv);
    if (c == 0) atomicAdd(&gcnt[b], 1.0f);
}

// ---- build cat = [hy | gsum/max(cnt,1)] -----------------------------------
__global__ void gatv2_build_cat(const float* __restrict__ hy,
                                const float* __restrict__ gsum,
                                const float* __restrict__ gcnt,
                                float* __restrict__ cat) {
    int t = blockIdx.x * blockDim.x + threadIdx.x;
    if (t >= N_GRAPH * D_CAT) return;
    int g = t / D_CAT, k = t - g * D_CAT;
    float v;
    if (k < D_ENC) v = hy[g * D_ENC + k];
    else           v = gsum[g * N_CHAN + (k - D_ENC)] / fmaxf(gcnt[g], 1.0f);
    cat[t] = v;
}

extern "C" void kernel_launch(void* const* d_in, const int* in_sizes, int n_in,
                              void* d_out, int out_size, void* d_ws, size_t ws_size,
                              hipStream_t stream) {
    const float* hy    = (const float*)d_in[0];
    const float* x     = (const float*)d_in[1];
    const int*   ei    = (const int*)  d_in[2];
    const int*   batch = (const int*)  d_in[3];
    const float* Wl    = (const float*)d_in[4];
    const float* bl    = (const float*)d_in[5];
    const float* Wr    = (const float*)d_in[6];
    const float* br    = (const float*)d_in[7];
    const float* att   = (const float*)d_in[8];
    const float* bgnn  = (const float*)d_in[9];
    const float* Wfc   = (const float*)d_in[10];
    const float* bfc   = (const float*)d_in[11];
    float* out = (float*)d_out;

    // ---- carve workspace (~131 MB) ----
    char* base = (char*)d_ws;
    size_t off = 0;
    auto carve = [&](size_t bytes) -> char* {
        char* p = base + off;
        off = (off + bytes + 255) & ~(size_t)255;
        return p;
    };
    float*    xl    = (float*)   carve((size_t)N_NODES * HC * 4);     // 51.2 MB
    float*    xr    = (float*)   carve((size_t)N_NODES * HC * 4);     // 51.2 MB
    float*    ebuf  = (float*)   carve((size_t)ET * N_HEAD * 4);      // 13.6 MB
    unsigned* menc  = (unsigned*)carve((size_t)N_NODES * N_HEAD * 4); // 0.8 MB
    float*    denom = (float*)   carve((size_t)N_NODES * N_HEAD * 4); // 0.8 MB
    float*    acc   = (float*)   carve((size_t)N_NODES * N_CHAN * 4); // 12.8 MB
    float*    gsum  = (float*)   carve((size_t)N_GRAPH * N_CHAN * 4);
    float*    gcnt  = (float*)   carve((size_t)N_GRAPH * 4);
    float*    cat   = (float*)   carve((size_t)N_GRAPH * D_CAT * 4);

    (void)hipMemsetAsync(menc,  0, (size_t)N_NODES * N_HEAD * 4, stream);
    (void)hipMemsetAsync(denom, 0, (size_t)N_NODES * N_HEAD * 4, stream);
    (void)hipMemsetAsync(acc,   0, (size_t)N_NODES * N_CHAN * 4, stream);
    (void)hipMemsetAsync(gsum,  0, (size_t)N_GRAPH * N_CHAN * 4, stream);
    (void)hipMemsetAsync(gcnt,  0, (size_t)N_GRAPH * 4, stream);

    // xl = x@Wl + bl ; xr = x@Wr + br    (tiles: 3125 x 16 = 50000 waves each)
    {
        int tilesN = HC / 16, totalTiles = (N_NODES / 16) * tilesN;
        int grid = (totalTiles + 7) / 8;
        gatv2_wmma_gemm<<<grid, 256, 0, stream>>>(x, Wl, bl, xl, D_IN, HC, tilesN, totalTiles);
        gatv2_wmma_gemm<<<grid, 256, 0, stream>>>(x, Wr, br, xr, D_IN, HC, tilesN, totalTiles);
    }

    // edge logits + segment max
    gatv2_edge_logits<<<(ET + 7) / 8, 256, 0, stream>>>(ei, xl, xr, att, ebuf, menc);
    // softmax denominators
    gatv2_edge_denom<<<(ET * N_HEAD + 255) / 256, 256, 0, stream>>>(ei, ebuf, menc, denom);
    // weighted scatter-aggregate
    gatv2_edge_aggregate<<<(ET + 7) / 8, 256, 0, stream>>>(ei, ebuf, menc, denom, xl, acc);
    // node finalize + graph pooling
    gatv2_node_pool<<<(N_NODES * N_CHAN + 255) / 256, 256, 0, stream>>>(acc, bgnn, batch, gsum, gcnt);
    // cat = [hy | hg]
    gatv2_build_cat<<<(N_GRAPH * D_CAT + 255) / 256, 256, 0, stream>>>(hy, gsum, gcnt, cat);

    // final FC: out[512,256] = cat[512,320] @ Wfc + bfc  (512 tiles)
    {
        int tilesN = D_ENC / 16, totalTiles = (N_GRAPH / 16) * tilesN;
        int grid = (totalTiles + 7) / 8;
        gatv2_wmma_gemm<<<grid, 256, 0, stream>>>(cat, Wfc, bfc, out, D_CAT, D_ENC, tilesN, totalTiles);
    }
}
